// SAE_24824910971121
// MI455X (gfx1250) — matrix operand order repaired and checked by reading, not compiled, not run on hardware
//
#include <hip/hip_runtime.h>

// Problem constants (match reference setup_inputs)
constexpr int Bn = 4096;   // batch
constexpr int Hn = 2048;   // hidden (GEMM K)
constexpr int Ln = 16384;  // latents
constexpr int Kn = 32;     // top-k

typedef __attribute__((ext_vector_type(16))) __bf16 v16bf;
typedef __attribute__((ext_vector_type(8)))  float  v8f;

constexpr int BM = 128, BN = 128, BK = 32;
constexpr int LDTF  = BK + 4;            // padded f32 LDS row stride = 36 dwords
constexpr int TILEB = BM * LDTF * 4;     // bytes per LDS tile buffer (18 KB)

#define ASYNC_B128(ldsoff, goff, base)                                   \
  asm volatile("global_load_async_to_lds_b128 %0, %1, %2"                \
               :: "v"(ldsoff), "v"(goff), "s"(base) : "memory")

// ---------------------------------------------------------------------------
// Kernel 1: encoder GEMM  z = relu(x @ W_enc^T), bf16 WMMA, async-LDS pipeline
// ---------------------------------------------------------------------------
__global__ __launch_bounds__(256) void sae_encoder(
    const float* __restrict__ x, const float* __restrict__ We,
    float* __restrict__ z)
{
  __shared__ float Af[2][BM * LDTF];   // 2 x 18 KB
  __shared__ float Bf[2][BN * LDTF];   // 2 x 18 KB

  const int t     = threadIdx.x;
  const int lane  = t & 31;
  const int wave  = t >> 5;
  const int lhalf = lane >> 4;
  const int lm    = lane & 15;

  const int rowBase = blockIdx.y * BM;
  const int colBase = blockIdx.x * BN;
  const int wm0 = (wave >> 1) * 32;
  const int wn0 = (wave & 1) * 64;

  // Async staging: wave w covers rows w*16..w*16+15; 4 issues per matrix,
  // each issue = 4 rows x 8 lanes x 16 B.
  const int rsub = lane >> 3;
  const int c4   = (lane & 7) * 4;

  unsigned goffX[4], goffW[4], loffA[4], loffB[4];
  #pragma unroll
  for (int q = 0; q < 4; ++q) {
    const int r = wave * 16 + q * 4 + rsub;
    goffX[q] = (unsigned)(((size_t)(rowBase + r) * Hn + c4) * sizeof(float));
    goffW[q] = (unsigned)(((size_t)(colBase + r) * Hn + c4) * sizeof(float));
    // low 32 bits of a generic __shared__ pointer = wave-relative LDS offset
    loffA[q] = (unsigned)(uintptr_t)&Af[0][r * LDTF + c4];
    loffB[q] = (unsigned)(uintptr_t)&Bf[0][r * LDTF + c4];
  }

  v8f acc[2][4];
  #pragma unroll
  for (int i = 0; i < 2; ++i)
    #pragma unroll
    for (int j = 0; j < 4; ++j)
      #pragma unroll
      for (int r = 0; r < 8; ++r) acc[i][j][r] = 0.0f;

  // prologue: stage tile 0 into buffer 0
  #pragma unroll
  for (int q = 0; q < 4; ++q) ASYNC_B128(loffA[q], goffX[q], x);
  #pragma unroll
  for (int q = 0; q < 4; ++q) ASYNC_B128(loffB[q], goffW[q], We);

  constexpr int NIT = Hn / BK;   // 64 K-steps
  for (int it = 0; it < NIT; ++it) {
    if (it + 1 < NIT) {
      const unsigned bo = ((it + 1) & 1) ? (unsigned)TILEB : 0u;
      const unsigned kb = (unsigned)((it + 1) * BK * sizeof(float));
      #pragma unroll
      for (int q = 0; q < 4; ++q) ASYNC_B128(loffA[q] + bo, goffX[q] + kb, x);
      #pragma unroll
      for (int q = 0; q < 4; ++q) ASYNC_B128(loffB[q] + bo, goffW[q] + kb, We);
      // async loads complete in order: <=8 outstanding => current tile landed
      asm volatile("s_wait_asynccnt 0x8" ::: "memory");
    } else {
      asm volatile("s_wait_asynccnt 0x0" ::: "memory");
    }
    __syncthreads();

    const float* Ab = Af[it & 1];
    const float* Bb = Bf[it & 1];

    // A 16x32 frag: lane<16 -> K 0-7 (elems 0-7), 16-23 (elems 8-15);
    //               lane>=16 -> K 8-15, 24-31.
    v16bf afr[2], bfr[4];
    #pragma unroll
    for (int i = 0; i < 2; ++i) {
      const float* ap = &Ab[(wm0 + i * 16 + lm) * LDTF + lhalf * 8];
      float pv[16];
      *(float4*)&pv[0]  = *(const float4*)(ap + 0);
      *(float4*)&pv[4]  = *(const float4*)(ap + 4);
      *(float4*)&pv[8]  = *(const float4*)(ap + 16);
      *(float4*)&pv[12] = *(const float4*)(ap + 20);
      #pragma unroll
      for (int e = 0; e < 16; ++e) afr[i][e] = (__bf16)pv[e];
    }
    // B 32x16 frag: lane n (0-15) = column n, K 0-7, 16-23; lane n+16: K 8-15, 24-31.
    #pragma unroll
    for (int j = 0; j < 4; ++j) {
      const float* bp = &Bb[(wn0 + j * 16 + lm) * LDTF + lhalf * 8];
      float pv[16];
      *(float4*)&pv[0]  = *(const float4*)(bp + 0);
      *(float4*)&pv[4]  = *(const float4*)(bp + 4);
      *(float4*)&pv[8]  = *(const float4*)(bp + 16);
      *(float4*)&pv[12] = *(const float4*)(bp + 20);
      #pragma unroll
      for (int e = 0; e < 16; ++e) bfr[j][e] = (__bf16)pv[e];
    }

    #pragma unroll
    for (int i = 0; i < 2; ++i)
      #pragma unroll
      for (int j = 0; j < 4; ++j)
        acc[i][j] = __builtin_amdgcn_wmma_f32_16x16x32_bf16(
            false, afr[i], false, bfr[j], (short)0, acc[i][j], false, false);

    __syncthreads();   // protect buffer (it&1) until all waves finished reading
  }

  // epilogue: fused ReLU store (C/D: VGPR r -> M=r / M=8+r by lane half)
  #pragma unroll
  for (int i = 0; i < 2; ++i) {
    const int gr0 = rowBase + wm0 + i * 16 + lhalf * 8;
    #pragma unroll
    for (int j = 0; j < 4; ++j) {
      const int gc = colBase + wn0 + j * 16 + lm;
      #pragma unroll
      for (int r = 0; r < 8; ++r)
        z[(size_t)(gr0 + r) * Ln + gc] = fmaxf(acc[i][j][r], 0.0f);
    }
  }
}

// ---------------------------------------------------------------------------
// Kernel 2: per-row top-K; row in registers, 32 argmax extractions.
// ---------------------------------------------------------------------------
__global__ __launch_bounds__(256) void sae_topk(
    float* __restrict__ z, float* __restrict__ vals, int* __restrict__ idxs)
{
  constexpr int EPT = Ln / 256;
  const int t = threadIdx.x;
  const int b = blockIdx.x;
  float* zrow = z + (size_t)b * Ln;

  float v[EPT];
  #pragma unroll
  for (int j = 0; j < EPT; ++j) v[j] = zrow[t + j * 256];

  __shared__ float rv[256];
  __shared__ int   ri[256];
  __shared__ float selv[Kn];
  __shared__ int   seli[Kn];

  for (int k = 0; k < Kn; ++k) {
    float bv = -1.0f; int bi = Ln;
    #pragma unroll
    for (int j = 0; j < EPT; ++j) {
      const int i = t + j * 256;
      if (v[j] > bv) { bv = v[j]; bi = i; }
    }
    rv[t] = bv; ri[t] = bi;
    __syncthreads();
    for (int s = 128; s > 0; s >>= 1) {
      if (t < s) {
        const float ov = rv[t + s]; const int oi = ri[t + s];
        if (ov > rv[t] || (ov == rv[t] && oi < ri[t])) { rv[t] = ov; ri[t] = oi; }
      }
      __syncthreads();
    }
    const int   besti = ri[0];
    const float bestv = rv[0];
    if (t == 0) { selv[k] = bestv; seli[k] = besti; }
    if ((besti & 255) == t) {
      const int jw = besti >> 8;
      #pragma unroll
      for (int j = 0; j < EPT; ++j) if (j == jw) v[j] = -1.0f;
    }
    __syncthreads();
  }

  #pragma unroll
  for (int j = 0; j < EPT; ++j) {
    const int i = t + j * 256;
    float out = 0.0f;
    if (v[j] == -1.0f) {
      for (int k = 0; k < Kn; ++k) if (seli[k] == i) out = selv[k];
    }
    zrow[i] = out;
  }
  if (t < Kn) { vals[b * Kn + t] = selv[t]; idxs[b * Kn + t] = seli[t]; }
}

// ---------------------------------------------------------------------------
// Kernel 3: W_dec [Hn, Ln] -> W_decT [Ln, Hn]
// ---------------------------------------------------------------------------
__global__ __launch_bounds__(256) void sae_transpose(
    const float* __restrict__ Wd, float* __restrict__ WdT)
{
  __shared__ float tile[32][33];
  const int l0 = blockIdx.x * 32;
  const int h0 = blockIdx.y * 32;
  const int tx = threadIdx.x;
  const int ty = threadIdx.y;
  #pragma unroll
  for (int r = ty; r < 32; r += 8)
    tile[r][tx] = Wd[(size_t)(h0 + r) * Ln + l0 + tx];
  __syncthreads();
  #pragma unroll
  for (int r = ty; r < 32; r += 8)
    WdT[(size_t)(l0 + r) * Hn + h0 + tx] = tile[tx][r];
}

// ---------------------------------------------------------------------------
// Kernel 4: sparse decoder  x_hat[b,:] = sum_k vals[b,k] * W_decT[idx[b,k], :]
// ---------------------------------------------------------------------------
__global__ __launch_bounds__(256) void sae_decoder(
    const float* __restrict__ WdT, const float* __restrict__ vals,
    const int* __restrict__ idxs, float* __restrict__ xhat)
{
  const int t = threadIdx.x;
  const int b = blockIdx.x;
  __shared__ float sv[Kn];
  __shared__ int   si[Kn];
  if (t < Kn) { sv[t] = vals[b * Kn + t]; si[t] = idxs[b * Kn + t]; }
  __syncthreads();

  float4 a0 = {0.f, 0.f, 0.f, 0.f};
  float4 a1 = {0.f, 0.f, 0.f, 0.f};
  #pragma unroll 4
  for (int k = 0; k < Kn; ++k) {
    const float w = sv[k];
    const float* row = WdT + (size_t)si[k] * Hn;
    const float4 r0 = *(const float4*)&row[t * 4];
    const float4 r1 = *(const float4*)&row[t * 4 + 1024];
    a0.x += w * r0.x; a0.y += w * r0.y; a0.z += w * r0.z; a0.w += w * r0.w;
    a1.x += w * r1.x; a1.y += w * r1.y; a1.z += w * r1.z; a1.w += w * r1.w;
  }
  float4* o = (float4*)&xhat[(size_t)b * Hn];
  o[t]       = a0;
  o[t + 256] = a1;
}

// ---------------------------------------------------------------------------
extern "C" void kernel_launch(void* const* d_in, const int* in_sizes, int n_in,
                              void* d_out, int out_size, void* d_ws, size_t ws_size,
                              hipStream_t stream) {
  const float* x  = (const float*)d_in[0];   // [Bn, Hn]
  const float* We = (const float*)d_in[1];   // [Ln, Hn]
  const float* Wd = (const float*)d_in[2];   // [Hn, Ln]

  float* xhat = (float*)d_out;               // [Bn, Hn]
  float* z    = xhat + (size_t)Bn * Hn;      // [Bn, Ln]

  float* vals = (float*)d_ws;                                    // 512 KB
  int*   idxs = (int*)((char*)d_ws + (size_t)Bn * Kn * sizeof(float));
  float* WdT  = (float*)((char*)d_ws + (size_t)(1u << 20));      // 128 MB

  sae_transpose<<<dim3(Ln / 32, Hn / 32), dim3(32, 8), 0, stream>>>(Wd, WdT);
  sae_encoder <<<dim3(Ln / BN, Bn / BM), 256, 0, stream>>>(x, We, z);
  sae_topk    <<<Bn, 256, 0, stream>>>(z, vals, idxs);
  sae_decoder <<<Bn, 256, 0, stream>>>(WdT, vals, idxs, xhat);
}
